// PairReprUpdate_62277025792456
// MI455X (gfx1250) — matrix operand order repaired and checked by reading, not compiled, not run on hardware
//
#include <hip/hip_runtime.h>
#include <hip/hip_bf16.h>

// ---------------- problem constants (from reference) ----------------
#define KNB   64     // neighbors per token
#define PDIM  128    // pair feature dim
#define HDIM  256    // transition hidden dim
#define TDIM  384    // token feature dim
#define EPSLN 1e-5f

// ---------------- LDS layout for pair_update (bytes) ----------------
// strides in elements, padded (+4 f32 / +8 bf16) to break bank conflicts
#define ZS  132   // zbuf  f32   [64][132]
#define AS  136   // abuf  bf16  [64][136]
#define HS  264   // hbuf  bf16  [64][264]
#define W1S 136   // w1t   bf16  [256][136]   (union with w2t)
#define W2S 264   // w2t   bf16  [128][264]
#define OFF_Z    0
#define OFF_A    33792      // 64*132*4
#define OFF_H    51200      // +64*136*2
#define OFF_W    84992      // +64*264*2
#define OFF_XP2  154624     // +max(256*136,128*264)*2 = +69632
#define OFF_TG   155136
#define OFF_TB   155648
#define OFF_PM   156160
#define OFF_IDX  156416
#define OFF_RED  156672
#define SMEM_BYTES 157184

typedef __attribute__((ext_vector_type(16))) __bf16 bf16x16;
typedef __attribute__((ext_vector_type(8)))  float  floatx8;

union Frag {
  bf16x16 v;
  uint4   q[2];
};

// native bf16 convert (v_cvt_*bf16_f32 on gfx1250), bit pattern as ushort
__device__ __forceinline__ unsigned short f2bf(float f) {
  __bf16 b = (__bf16)f;
  return __builtin_bit_cast(unsigned short, b);
}
// pack two floats -> 2x bf16 in a dword (lo in [15:0], hi in [31:16])
__device__ __forceinline__ unsigned int pack_bf2(float lo, float hi) {
  return (unsigned int)f2bf(lo) | ((unsigned int)f2bf(hi) << 16);
}

// A fragment (16-bit A 16x32 layout): two 16B chunks at K = koff.. and 16+koff..
__device__ __forceinline__ bf16x16 load_frag_a(const unsigned short* p) {
  Frag f;
  f.q[0] = *reinterpret_cast<const uint4*>(p);
  f.q[1] = *reinterpret_cast<const uint4*>(p + 16);
  return f.v;
}
// B fragment (16-bit B 32x16 layout): two 16B chunks at K = koff.. and 16+koff..
__device__ __forceinline__ bf16x16 load_frag_b(const unsigned short* p) {
  Frag f;
  f.q[0] = *reinterpret_cast<const uint4*>(p);
  f.q[1] = *reinterpret_cast<const uint4*>(p + 16);
  return f.v;
}

// =====================================================================
// Kernel 1: proj[n,:] = LN(x[n,:]*mask[n]) @ Wx    (4096 x 256, tiny)
// =====================================================================
__global__ __launch_bounds__(128)
void ln_proj_kernel(const float* __restrict__ x,
                    const unsigned char* __restrict__ mask,
                    const float* __restrict__ ln_g,
                    const float* __restrict__ ln_b,
                    const float* __restrict__ Wx,
                    float* __restrict__ proj) {
  __shared__ float xs[TDIM];
  __shared__ float red[128];
  const int n   = blockIdx.x;
  const int tid = threadIdx.x;           // 128 threads
  const float mf = mask[n] ? 1.f : 0.f;

  float s = 0.f;
  for (int k = tid; k < TDIM; k += 128) {
    float v = x[(size_t)n * TDIM + k] * mf;
    xs[k] = v; s += v;
  }
  red[tid] = s; __syncthreads();
  for (int st = 64; st > 0; st >>= 1) { if (tid < st) red[tid] += red[tid + st]; __syncthreads(); }
  const float mu = red[0] * (1.f / TDIM);
  __syncthreads();

  float vs = 0.f;
  for (int k = tid; k < TDIM; k += 128) { float d = xs[k] - mu; vs += d * d; }
  red[tid] = vs; __syncthreads();
  for (int st = 64; st > 0; st >>= 1) { if (tid < st) red[tid] += red[tid + st]; __syncthreads(); }
  const float rs = rsqrtf(red[0] * (1.f / TDIM) + EPSLN);
  __syncthreads();

  for (int k = tid; k < TDIM; k += 128)
    xs[k] = (xs[k] - mu) * rs * ln_g[k] + ln_b[k];
  __syncthreads();

  for (int c = tid; c < 2 * PDIM; c += 128) {
    float acc = 0.f;
    for (int k = 0; k < TDIM; ++k)
      acc = fmaf(xs[k], Wx[k * 2 * PDIM + c], acc);
    proj[(size_t)n * 2 * PDIM + c] = acc;
  }
}

// =====================================================================
// Kernel 2: pair update + transition.  One block per token n.
// 128 threads = 4 waves; wave w owns rows (k-slots) 16w..16w+15.
// =====================================================================
__global__ __launch_bounds__(128)
void pair_update_kernel(const float* __restrict__ pair_rep,
                        const unsigned char* __restrict__ mask,
                        const int* __restrict__ nbr_idx,
                        const unsigned char* __restrict__ slot_valid,
                        const float* __restrict__ t_g,
                        const float* __restrict__ t_b,
                        const float* __restrict__ W1,
                        const float* __restrict__ b1,
                        const float* __restrict__ W2,
                        const float* __restrict__ b2,
                        const float* __restrict__ proj,
                        float* __restrict__ out) {
  extern __shared__ char smem[];
  float*          zbuf   = (float*)(smem + OFF_Z);
  unsigned short* abuf   = (unsigned short*)(smem + OFF_A);
  unsigned short* hbuf   = (unsigned short*)(smem + OFF_H);
  unsigned short* wbuf   = (unsigned short*)(smem + OFF_W);
  float*          xp2buf = (float*)(smem + OFF_XP2);
  float*          tgbuf  = (float*)(smem + OFF_TG);
  float*          tbbuf  = (float*)(smem + OFF_TB);
  float*          pmaskb = (float*)(smem + OFF_PM);
  int*            idxb   = (int*)(smem + OFF_IDX);
  float*          redb   = (float*)(smem + OFF_RED);

  const int n    = blockIdx.x;
  const int tid  = threadIdx.x;     // 0..127
  const int lane = tid & 31;
  const int wv   = tid >> 5;

  // ---- Phase 0: broadcast row data ----
  xp2buf[tid] = proj[(size_t)n * 256 + 128 + tid];
  tgbuf[tid]  = t_g[tid];
  tbbuf[tid]  = t_b[tid];
  if (tid < KNB) {
    const int gi = nbr_idx[n * KNB + tid];
    idxb[tid] = gi;
    pmaskb[tid] = (mask[n] && mask[gi] && slot_valid[n * KNB + tid]) ? 1.f : 0.f;
  }
  __syncthreads();

  // ---- Phase 1: z = (pair_rep + xp2 + gather(xp1)) * pmask; LN(z) -> abuf (bf16) ----
  {
    const int rr = tid >> 1, half = tid & 1, c0 = half * 64;
    const float pm = pmaskb[rr];
    const int   gi = idxb[rr];
    const float4* prp4 = (const float4*)(pair_rep + ((size_t)(n * KNB + rr)) * PDIM + c0);
    const float4* x1p4 = (const float4*)(proj + (size_t)gi * 256 + c0);
    const float4* xp24 = (const float4*)(xp2buf + c0);
    float4* zr4 = (float4*)(zbuf + rr * ZS + c0);
    float s = 0.f;
    for (int j = 0; j < 16; ++j) {
      float4 a = prp4[j], g = x1p4[j], c = xp24[j], v;
      v.x = (a.x + g.x + c.x) * pm;  v.y = (a.y + g.y + c.y) * pm;
      v.z = (a.z + g.z + c.z) * pm;  v.w = (a.w + g.w + c.w) * pm;
      zr4[j] = v;
      s += v.x + v.y + v.z + v.w;
    }
    redb[tid] = s;
    __syncthreads();
    const float mu = (redb[rr * 2] + redb[rr * 2 + 1]) * (1.f / PDIM);
    __syncthreads();
    const float* zr = zbuf + rr * ZS + c0;
    float vs = 0.f;
    for (int j = 0; j < 64; ++j) { float d = zr[j] - mu; vs += d * d; }
    redb[tid] = vs;
    __syncthreads();
    const float var = (redb[rr * 2] + redb[rr * 2 + 1]) * (1.f / PDIM);
    const float rsv = rsqrtf(var + EPSLN);
    // packed 2x bf16 stores (c0 is even, abuf rows 4B aligned)
    unsigned int* ar2 = (unsigned int*)(abuf + rr * AS + c0);
    for (int j = 0; j < 32; ++j) {
      const int c = c0 + 2 * j;
      const float a0 = (zr[2 * j]     - mu) * rsv * tgbuf[c]     + tbbuf[c];
      const float a1 = (zr[2 * j + 1] - mu) * rsv * tgbuf[c + 1] + tbbuf[c + 1];
      ar2[j] = pack_bf2(a0, a1);
    }
  }
  __syncthreads();

  // ---- Phase 2: stage W1^T as bf16: wbuf[nn*W1S + k] = W1[k][nn] ----
  // thread owns (k,k+1) pair for one column -> packed b32 LDS store,
  // both global reads coalesced across threads (consecutive nn).
  for (int e = tid; e < (PDIM / 2) * HDIM; e += 128) {
    const int kp = e >> 8;      // 0..63  -> k = 2*kp
    const int nn = e & 255;     // 0..255 (coalesced over threads)
    const float w0 = W1[(2 * kp)     * HDIM + nn];
    const float w1 = W1[(2 * kp + 1) * HDIM + nn];
    ((unsigned int*)(wbuf + nn * W1S))[kp] = pack_bf2(w0, w1);
  }
  __syncthreads();

  // ---- Phase 3: GEMM1  h = relu(LN(z) @ W1 + b1) -> hbuf bf16 ----
  {
    const int rbase = wv * 16;
    const int m     = lane & 15;
    const int hihf  = lane >> 4;        // 0 or 1
    const int koffA = hihf * 8;
    const int koffB = hihf * 8;
    bf16x16 fa[4];
    const unsigned short* arow = abuf + (rbase + m) * AS;
    for (int kk = 0; kk < 4; ++kk)
      fa[kk] = load_frag_a(arow + kk * 32 + koffA);
    for (int nt = 0; nt < 16; ++nt) {
      const unsigned short* bcol = wbuf + (nt * 16 + m) * W1S;
      floatx8 acc = {0.f, 0.f, 0.f, 0.f, 0.f, 0.f, 0.f, 0.f};
      for (int kk = 0; kk < 4; ++kk) {
        bf16x16 fb = load_frag_b(bcol + kk * 32 + koffB);
        acc = __builtin_amdgcn_wmma_f32_16x16x32_bf16(false, fa[kk], false, fb,
                                                      (short)0, acc, false, false);
      }
      const int   ncol = nt * 16 + m;
      const float bias = b1[ncol];
      const int   rofs = rbase + hihf * 8;
      for (int r = 0; r < 8; ++r) {
        const float hv = fmaxf(acc[r] + bias, 0.f);   // v_max_num_f32
        hbuf[(rofs + r) * HS + ncol] = f2bf(hv);
      }
    }
  }
  __syncthreads();

  // ---- Phase 4: stage W2^T as bf16 (overwrites W1 staging) ----
  // thread owns (h,h+1) pair for one column p -> packed b32 LDS store.
  for (int e = tid; e < (HDIM / 2) * PDIM; e += 128) {
    const int hp = e >> 7;      // 0..127 -> h = 2*hp
    const int p  = e & 127;     // 0..127 (coalesced over threads)
    const float w0 = W2[(2 * hp)     * PDIM + p];
    const float w1 = W2[(2 * hp + 1) * PDIM + p];
    ((unsigned int*)(wbuf + p * W2S))[hp] = pack_bf2(w0, w1);
  }
  __syncthreads();

  // ---- Phase 5: GEMM2 + epilogue: out = z + (h@W2 + b2) * pmask ----
  {
    const int rbase = wv * 16;
    const int m     = lane & 15;
    const int hihf  = lane >> 4;
    const int koffA = hihf * 8;
    const int koffB = hihf * 8;
    bf16x16 ha[8];
    const unsigned short* hrow = hbuf + (rbase + m) * HS;
    for (int kk = 0; kk < 8; ++kk)
      ha[kk] = load_frag_a(hrow + kk * 32 + koffA);
    for (int nt = 0; nt < 8; ++nt) {
      const unsigned short* bcol = wbuf + (nt * 16 + m) * W2S;
      floatx8 acc = {0.f, 0.f, 0.f, 0.f, 0.f, 0.f, 0.f, 0.f};
      for (int kk = 0; kk < 8; ++kk) {
        bf16x16 fb = load_frag_b(bcol + kk * 32 + koffB);
        acc = __builtin_amdgcn_wmma_f32_16x16x32_bf16(false, ha[kk], false, fb,
                                                      (short)0, acc, false, false);
      }
      const int   pcol = nt * 16 + m;
      const float bias = b2[pcol];
      const int   rofs = rbase + hihf * 8;
      for (int r = 0; r < 8; ++r) {
        const int row = rofs + r;
        const float val = zbuf[row * ZS + pcol] + (acc[r] + bias) * pmaskb[row];
        out[((size_t)(n * KNB + row)) * PDIM + pcol] = val;
      }
    }
  }
}

// =====================================================================
extern "C" void kernel_launch(void* const* d_in, const int* in_sizes, int n_in,
                              void* d_out, int out_size, void* d_ws, size_t ws_size,
                              hipStream_t stream) {
  const float*         x    = (const float*)d_in[0];
  const float*         prep = (const float*)d_in[1];
  const unsigned char* mask = (const unsigned char*)d_in[2];
  const int*           nbr  = (const int*)d_in[3];
  const unsigned char* slot = (const unsigned char*)d_in[4];
  const float*         ln_g = (const float*)d_in[5];
  const float*         ln_b = (const float*)d_in[6];
  const float*         Wx   = (const float*)d_in[7];
  const float*         t_g  = (const float*)d_in[8];
  const float*         t_b  = (const float*)d_in[9];
  const float*         W1   = (const float*)d_in[10];
  const float*         b1   = (const float*)d_in[11];
  const float*         W2   = (const float*)d_in[12];
  const float*         b2   = (const float*)d_in[13];
  float* out  = (float*)d_out;
  float* proj = (float*)d_ws;                  // N x 256 f32 scratch

  const int N = in_sizes[2];                   // mask has B*N elements, B=1

  (void)hipFuncSetAttribute((const void*)pair_update_kernel,
                            hipFuncAttributeMaxDynamicSharedMemorySize, SMEM_BYTES);

  hipLaunchKernelGGL(ln_proj_kernel, dim3(N), dim3(128), 0, stream,
                     x, mask, ln_g, ln_b, Wx, proj);
  hipLaunchKernelGGL(pair_update_kernel, dim3(N), dim3(128), SMEM_BYTES, stream,
                     prep, mask, nbr, slot, t_g, t_b, W1, b1, W2, b2, proj, out);
}
